// GAU_38001870635641
// MI455X (gfx1250) — matrix operand order repaired and checked by reading, not compiled, not run on hardware
//
#include <hip/hip_runtime.h>
#include <hip/hip_bf16.h>

typedef __bf16 v16bf __attribute__((ext_vector_type(16)));
typedef float  v8f   __attribute__((ext_vector_type(8)));
typedef float  f32x4 __attribute__((ext_vector_type(4)));

#define DIM_    768
#define QKD_    128
#define HID_    1536
#define BB_     4
#define SS_     2048
#define MTOT_   (BB_ * SS_)   // 8192

// ---------- helpers ----------
__device__ __forceinline__ __bf16 f2bf(float f) {
    unsigned u = __builtin_bit_cast(unsigned, f);
    unsigned r = u + 0x7FFFu + ((u >> 16) & 1u);
    return __builtin_bit_cast(__bf16, (unsigned short)(r >> 16));
}

__device__ __forceinline__ float silu_f(float x) {
    return x / (1.0f + __expf(-x));
}

// A fragment (16x32, M x K): lane m = lane&15, half = lane>>4.
// elements j=0..7  -> K = half*8 + j
// elements j=8..15 -> K = 16 + half*8 + (j-8)
__device__ __forceinline__ v16bf frag_a(const __bf16* base, int stride) {
    int lane = threadIdx.x & 31;
    const __bf16* p = base + (lane & 15) * stride + ((lane >> 4) << 3);
    union { v16bf v; f32x4 q[2]; } u;
    u.q[0] = *(const f32x4*)(p);
    u.q[1] = *(const f32x4*)(p + 16);
    return u.v;
}

// B fragment (32x16, K x N), read from a [N][K] (K-major per N row) layout.
// lane n = lane&15, half = lane>>4; elements j=0..7 -> K = half*8 + j, j=8..15 -> K = 16 + half*8 + (j-8)
__device__ __forceinline__ v16bf frag_b(const __bf16* base, int stride) {
    int lane = threadIdx.x & 31;
    const __bf16* p = base + (lane & 15) * stride + ((lane >> 4) << 3);
    union { v16bf v; f32x4 q[2]; } u;
    u.q[0] = *(const f32x4*)(p);
    u.q[1] = *(const f32x4*)(p + 16);
    return u.v;
}

__device__ __forceinline__ v8f wmma_bf16(v16bf a, v16bf b, v8f c) {
    return __builtin_amdgcn_wmma_f32_16x16x32_bf16(false, a, false, b, (short)0, c,
                                                   false, false);
}

// gfx1250 async global -> LDS DMA (tracked by ASYNCcnt).
// lds_off: LDS byte offset (low 32 bits of flat shared address per aperture rule).
__device__ __forceinline__ void async_copy_b128(unsigned lds_off,
                                                const __bf16* gaddr) {
    asm volatile("global_load_async_to_lds_b128 %0, %1, off"
                 :: "v"(lds_off), "v"(gaddr)
                 : "memory");
}
__device__ __forceinline__ void wait_async0() {
    asm volatile("s_wait_asynccnt 0" ::: "memory");
}

// ---------- fp32 -> bf16 convert ----------
__global__ void cvt_kernel(const float* __restrict__ in, __bf16* __restrict__ out, long n) {
    long i = (long)blockIdx.x * blockDim.x + threadIdx.x;
    long step = (long)gridDim.x * blockDim.x;
    for (; i < n; i += step) out[i] = f2bf(in[i]);
}

// ---------- LayerNorm -> bf16 ----------
__global__ __launch_bounds__(256) void ln_kernel(const float* __restrict__ x,
                                                 const float* __restrict__ g,
                                                 const float* __restrict__ b,
                                                 __bf16* __restrict__ out) {
    int row = blockIdx.x;
    const float* xr = x + (long)row * DIM_;
    int tid = threadIdx.x;
    float v[3];
    float s = 0.f, s2 = 0.f;
#pragma unroll
    for (int i = 0; i < 3; ++i) {
        v[i] = xr[tid + i * 256];
        s += v[i];
        s2 += v[i] * v[i];
    }
#pragma unroll
    for (int o = 16; o > 0; o >>= 1) {
        s += __shfl_down(s, o);
        s2 += __shfl_down(s2, o);
    }
    __shared__ float red[16];
    __shared__ float mv[2];
    int w = tid >> 5, lane = tid & 31;
    if (lane == 0) { red[w] = s; red[8 + w] = s2; }
    __syncthreads();
    if (tid == 0) {
        float ts = 0.f, ts2 = 0.f;
        for (int i = 0; i < 8; ++i) { ts += red[i]; ts2 += red[8 + i]; }
        float mu = ts * (1.0f / DIM_);
        float var = ts2 * (1.0f / DIM_) - mu * mu;
        mv[0] = mu;
        mv[1] = rsqrtf(var + 1e-5f);
    }
    __syncthreads();
    float mu = mv[0], rs = mv[1];
#pragma unroll
    for (int i = 0; i < 3; ++i) {
        int c = tid + i * 256;
        out[(long)row * DIM_ + c] = f2bf((v[i] - mu) * rs * g[c] + b[c]);
    }
}

// ---------- generic LDS-tiled bf16 WMMA GEMM with fused epilogues ----------
// Block tile 64(M) x 128(N), 8 waves each computing 32x32, K step 32.
// A tile staged via GLOBAL_LOAD_ASYNC_TO_LDS_B128 (ASYNCcnt path);
// B tile staged transposed via regular loads + ds stores.
// EPI 0: HIDDEN  (bias + SiLU; cols<HID -> v bf16, cols>=HID -> gate f32)
// EPI 1: QK      (bias + SiLU; q = c*f0+f1 -> ob0, k = c*f2+f3 -> ob1)
// EPI 2: AV      (c *= gate[f0]; -> ob0 bf16; batched via blockIdx.z)
// EPI 3: OUT     (c += bias + resid; -> outf f32)
template <int EPI>
__global__ __launch_bounds__(256) void gemm_kernel(
    const __bf16* __restrict__ A, const __bf16* __restrict__ B, int N, int K,
    long zA, long zB, const float* __restrict__ bias, const float* __restrict__ f0,
    const float* __restrict__ f1, const float* __restrict__ f2,
    const float* __restrict__ f3, long zf0, __bf16* __restrict__ ob0,
    __bf16* __restrict__ ob1, float* __restrict__ of0, long zob0,
    const float* __restrict__ resid, float* __restrict__ outf) {
    __shared__ __bf16 sA[64 * 32];
    __shared__ __bf16 sB[128 * 32];

    int tid = threadIdx.x;
    int w = tid >> 5, lane = tid & 31;
    int wm = w >> 2, wn = w & 3;
    int gm0 = blockIdx.y * 64;
    int n0 = blockIdx.x * 128;
    long z = blockIdx.z;

    const __bf16* Ab = A + z * zA;
    const __bf16* Bb = B + z * zB;

    // per-thread A-tile copy slot (layout-preserving: [m][k], stride 32)
    int am = tid >> 2, akp = (tid & 3) << 3;
    unsigned aslot = (unsigned)(uintptr_t)(sA + am * 32 + akp);

    v8f acc[2][2] = {};

    for (int k0 = 0; k0 < K; k0 += 32) {
        // stage A tile 64x32 via async DMA to LDS (gfx1250)
        async_copy_b128(aslot, Ab + (long)(gm0 + am) * K + k0 + akp);
        // stage B tile transposed: sB[n][k], 128x32
        for (int c = tid; c < 512; c += 256) {
            int kk = c >> 4, ng = (c & 15) << 3;
            union { f32x4 q; __bf16 e[8]; } u;
            u.q = *(const f32x4*)(Bb + (long)(k0 + kk) * N + n0 + ng);
#pragma unroll
            for (int i = 0; i < 8; ++i) sB[(ng + i) * 32 + kk] = u.e[i];
        }
        if (k0 + 32 < K)  // gfx1250 global_prefetch of next B tile
            __builtin_prefetch(Bb + (long)(k0 + 32 + (tid >> 4)) * N + n0, 0, 0);
        wait_async0();
        __syncthreads();

        v16bf bfrag[2];
#pragma unroll
        for (int j = 0; j < 2; ++j)
            bfrag[j] = frag_b(sB + (wn * 32 + j * 16) * 32, 32);
#pragma unroll
        for (int i = 0; i < 2; ++i) {
            v16bf a = frag_a(sA + (wm * 32 + i * 16) * 32, 32);
#pragma unroll
            for (int j = 0; j < 2; ++j)
                acc[i][j] = wmma_bf16(a, bfrag[j], acc[i][j]);
        }
        __syncthreads();
    }

    int half = lane >> 4, cn = lane & 15;
#pragma unroll
    for (int i = 0; i < 2; ++i)
#pragma unroll
        for (int j = 0; j < 2; ++j)
#pragma unroll
            for (int r = 0; r < 8; ++r) {
                int gm = gm0 + wm * 32 + i * 16 + half * 8 + r;
                int gn = n0 + wn * 32 + j * 16 + cn;
                float c = acc[i][j][r];
                if constexpr (EPI == 0) {
                    c = silu_f(c + bias[gn]);
                    if (gn < HID_)
                        ob0[(long)gm * HID_ + gn] = f2bf(c);
                    else
                        of0[(long)gm * HID_ + (gn - HID_)] = c;
                } else if constexpr (EPI == 1) {
                    c = silu_f(c + bias[gn]);
                    ob0[(long)gm * QKD_ + gn] = f2bf(c * f0[gn] + f1[gn]);
                    ob1[(long)gm * QKD_ + gn] = f2bf(c * f2[gn] + f3[gn]);
                } else if constexpr (EPI == 2) {
                    const float* gatep = f0 + z * zf0;
                    __bf16* gavp = ob0 + z * zob0;
                    c *= gatep[(long)gm * HID_ + gn];
                    gavp[(long)gm * HID_ + gn] = f2bf(c);
                } else {  // EPI == 3
                    c += bias[gn] + resid[(long)gm * DIM_ + gn];
                    outf[(long)gm * DIM_ + gn] = c;
                }
            }
}

// ---------- relu^2 attention scores: attn = relu(q k^T / S)^2 (bf16) ----------
__global__ __launch_bounds__(256) void score_kernel(const __bf16* __restrict__ q,
                                                    const __bf16* __restrict__ k,
                                                    const unsigned char* __restrict__ mask,
                                                    __bf16* __restrict__ attn,
                                                    float inv_s) {
    long z = blockIdx.z;
    const __bf16* Q = q + z * (long)SS_ * QKD_;
    const __bf16* Km = k + z * (long)SS_ * QKD_;
    const unsigned char* mz = mask + z * SS_;
    __bf16* O = attn + z * (long)SS_ * SS_;

    int w = threadIdx.x >> 5, lane = threadIdx.x & 31;
    int m0 = blockIdx.y * 32;
    int n0 = blockIdx.x * 256 + w * 32;

    v8f acc[2][2] = {};
#pragma unroll
    for (int k0 = 0; k0 < QKD_; k0 += 32) {
        v16bf a0 = frag_a(Q + (long)m0 * QKD_ + k0, QKD_);
        v16bf a1 = frag_a(Q + (long)(m0 + 16) * QKD_ + k0, QKD_);
        v16bf b0 = frag_b(Km + (long)n0 * QKD_ + k0, QKD_);
        v16bf b1 = frag_b(Km + (long)(n0 + 16) * QKD_ + k0, QKD_);
        acc[0][0] = wmma_bf16(a0, b0, acc[0][0]);
        acc[0][1] = wmma_bf16(a0, b1, acc[0][1]);
        acc[1][0] = wmma_bf16(a1, b0, acc[1][0]);
        acc[1][1] = wmma_bf16(a1, b1, acc[1][1]);
    }

    int half = lane >> 4, cn = lane & 15;
#pragma unroll
    for (int i = 0; i < 2; ++i)
#pragma unroll
        for (int j = 0; j < 2; ++j)
#pragma unroll
            for (int r = 0; r < 8; ++r) {
                int gm = m0 + i * 16 + half * 8 + r;
                int gn = n0 + j * 16 + cn;
                float s = acc[i][j][r] * inv_s;
                s = fmaxf(s, 0.0f);
                s *= s;
                if (!mz[gn]) s = 0.0f;
                O[(long)gm * SS_ + gn] = f2bf(s);
            }
}

// ---------- host launcher ----------
extern "C" void kernel_launch(void* const* d_in, const int* in_sizes, int n_in,
                              void* d_out, int out_size, void* d_ws, size_t ws_size,
                              hipStream_t stream) {
    (void)in_sizes; (void)n_in; (void)out_size; (void)ws_size;
    const float* hidden      = (const float*)d_in[0];
    const unsigned char* msk = (const unsigned char*)d_in[1];
    const float* ln_g        = (const float*)d_in[2];
    const float* ln_b        = (const float*)d_in[3];
    const float* w_hidden    = (const float*)d_in[4];
    const float* b_hidden    = (const float*)d_in[5];
    const float* w_qk        = (const float*)d_in[6];
    const float* b_qk        = (const float*)d_in[7];
    const float* os_g        = (const float*)d_in[8];   // [2,128]
    const float* os_b        = (const float*)d_in[9];   // [2,128]
    const float* w_out       = (const float*)d_in[10];
    const float* b_out       = (const float*)d_in[11];
    float* out               = (float*)d_out;

    char* ws = (char*)d_ws;
    size_t off = 0;
    auto alloc = [&](size_t bytes) -> char* {
        char* p = ws + off;
        off += (bytes + 255) & ~(size_t)255;
        return p;
    };
    __bf16* normed = (__bf16*)alloc((size_t)MTOT_ * DIM_ * 2);
    __bf16* whb    = (__bf16*)alloc((size_t)DIM_ * 2 * HID_ * 2);
    __bf16* wqkb   = (__bf16*)alloc((size_t)DIM_ * QKD_ * 2);
    __bf16* woutb  = (__bf16*)alloc((size_t)HID_ * DIM_ * 2);
    __bf16* vb     = (__bf16*)alloc((size_t)MTOT_ * HID_ * 2);
    float*  gate   = (float*) alloc((size_t)MTOT_ * HID_ * 4);
    __bf16* qb     = (__bf16*)alloc((size_t)MTOT_ * QKD_ * 2);
    __bf16* kb     = (__bf16*)alloc((size_t)MTOT_ * QKD_ * 2);
    __bf16* attnb  = (__bf16*)alloc((size_t)BB_ * SS_ * SS_ * 2);
    __bf16* gavb   = (__bf16*)alloc((size_t)MTOT_ * HID_ * 2);

    // 1) weights -> bf16
    cvt_kernel<<<dim3(1024), 256, 0, stream>>>(w_hidden, whb, (long)DIM_ * 2 * HID_);
    cvt_kernel<<<dim3(128), 256, 0, stream>>>(w_qk, wqkb, (long)DIM_ * QKD_);
    cvt_kernel<<<dim3(512), 256, 0, stream>>>(w_out, woutb, (long)HID_ * DIM_);

    // 2) layernorm
    ln_kernel<<<dim3(MTOT_), 256, 0, stream>>>(hidden, ln_g, ln_b, normed);

    // 3) hidden GEMM: [8192,768] x [768,3072] -> SiLU -> v(bf16), gate(f32)
    gemm_kernel<0><<<dim3((2 * HID_) / 128, MTOT_ / 64, 1), 256, 0, stream>>>(
        normed, whb, 2 * HID_, DIM_, 0, 0, b_hidden, nullptr, nullptr, nullptr,
        nullptr, 0, vb, nullptr, gate, 0, nullptr, nullptr);

    // 4) qk GEMM: [8192,768] x [768,128] -> SiLU -> q,k (bf16)
    gemm_kernel<1><<<dim3(QKD_ / 128, MTOT_ / 64, 1), 256, 0, stream>>>(
        normed, wqkb, QKD_, DIM_, 0, 0, b_qk, os_g, os_b, os_g + QKD_, os_b + QKD_,
        0, qb, kb, nullptr, 0, nullptr, nullptr);

    // 5) attention scores: relu(q k^T / S)^2 -> bf16, masked
    score_kernel<<<dim3(SS_ / 256, SS_ / 32, BB_), 256, 0, stream>>>(
        qb, kb, msk, attnb, 1.0f / (float)SS_);

    // 6) AV GEMM (batched): attn[z] x v[z] * gate -> gav (bf16)
    gemm_kernel<2><<<dim3(HID_ / 128, SS_ / 64, BB_), 256, 0, stream>>>(
        attnb, vb, HID_, SS_, (long)SS_ * SS_, (long)SS_ * HID_, nullptr, gate,
        nullptr, nullptr, nullptr, (long)SS_ * HID_, gavb, nullptr, nullptr,
        (long)SS_ * HID_, nullptr, nullptr);

    // 7) output GEMM: [8192,1536] x [1536,768] + b_out + residual -> fp32 out
    gemm_kernel<3><<<dim3(DIM_ / 128, MTOT_ / 64, 1), 256, 0, stream>>>(
        gavb, woutb, DIM_, HID_, 0, 0, b_out, nullptr, nullptr, nullptr, nullptr,
        0, nullptr, nullptr, nullptr, 0, hidden, out);
}
